// DFHEncoder_75488345194661
// MI455X (gfx1250) — compile-verified, matrix operand order checked by reading, not run on hardware
//
#include <hip/hip_runtime.h>
#include <hip/hip_bf16.h>

// ---------------------------------------------------------------------------
// MI455X (gfx1250) implementation, round 6.
//   * GEMM tiles staged by the Tensor Data Mover: tensor_load_to_lds moves
//     bf16 2D tiles global->LDS (row pad 64B+16B == D# pad_interval=16dw,
//     pad_amount=4dw), issued by wave 0 per K-step, s_wait_tensorcnt before
//     the ping-pong barrier. No per-thread staging, no VGPR round trip.
//   * All GEMM operands pre-converted once to bf16 (B transposed to [N,K]);
//     activations carry a 2-row zero halo padded to the 128-row block grid
//     -> guard-free A tiles (conv tap == plain row offset).
//   * 4x v_wmma_f32_16x16x32_bf16 per K-step (128x64 block, 32x32 per wave),
//     fragments are contiguous 16B LDS chunks (ds_load_b128).
//   * GCN propagation = memory-bound gather/scatter (atomicAdd f32).
//   * Attention algebraically reduced: emb = ((sum_i w_i feats_i)@Wv + bv),
//     score_i = feats_i.(Wk qa) + qa.bk.
// ---------------------------------------------------------------------------

typedef __attribute__((ext_vector_type(16))) __bf16 bf16x16;
typedef __attribute__((ext_vector_type(8)))  float  f32x8;
typedef __attribute__((ext_vector_type(8)))  unsigned short u16x8;
typedef __attribute__((ext_vector_type(4)))  unsigned int v4u32;
typedef __attribute__((ext_vector_type(4)))  int v4i32;
typedef __attribute__((ext_vector_type(8)))  int v8i32;

union BF16Frag { bf16x16 v; u16x8 h[2]; };

__device__ __forceinline__ unsigned int f2bfbits(float f) {
  unsigned int u = __float_as_uint(f);
  u += 0x7FFFu + ((u >> 16) & 1u);          // round-to-nearest-even
  return u >> 16;
}

// Issue a TDM 2D bf16 tile load: tileRows x 32 elements, row stride
// `strideElems` (elements), LDS rows padded 64B data + 16B pad.
__device__ __forceinline__ void tdm_tile_load(
    const unsigned short* gbase, unsigned int lds_addr,
    int strideElems, int tileRows)
{
  unsigned long ga = (unsigned long)gbase;
  v4u32 g0;
  g0[0] = 1u;                                             // count=1, user D#
  g0[1] = lds_addr;                                       // [63:32] lds_addr
  g0[2] = (unsigned int)(ga & 0xFFFFFFFFu);               // global_addr lo
  g0[3] = (unsigned int)((ga >> 32) & 0x01FFFFFFu)        // global_addr hi
        | 0x80000000u;                                    // type=2 ("image")
  v8i32 g1;
  unsigned int td0 = (unsigned int)strideElems;           // tensor_dim0
  unsigned int td1 = 1u << 20;                            // tensor_dim1 (big)
  g1[0] = (int)((1u << 16)                                // data_size: 2B
        |       (1u << 20)                                // pad_enable
        |       (3u << 22)                                // pad_interval: 16dw
        |       (3u << 25));                              // pad_amount: 4dw
  g1[1] = (int)((td0 & 0xFFFFu) << 16);                   // dim0[15:0] @ [63:48]
  g1[2] = (int)(((td0 >> 16) & 0xFFFFu) | ((td1 & 0xFFFFu) << 16));
  g1[3] = (int)(((td1 >> 16) & 0xFFFFu) | (32u << 16));   // tile_dim0 = 32
  g1[4] = (int)((unsigned)tileRows & 0xFFFFu);            // tile_dim1
  g1[5] = (int)td0;                                       // dim0_stride lo32
  g1[6] = 0;
  g1[7] = 0;
  v4i32 z4 = {};
  v8i32 z8 = {};
  __builtin_amdgcn_tensor_load_to_lds(g0, g1, z4, z4, z8, 0);
}

// ---------------------------------------------------------------------------
// C[M,N] = act(A' @ B + bias). A: halo'd bf16 [rowsH, lda], logical row m at
// halo row m+2 (outside rows zero). B: bf16 [N, K] row-major (transposed).
// N % 64 == 0, K % 32 == 0, lda % 32 == 0.
// CONV: lda = Cin, K = 5*Cin, tap-major im2col, per-K-tile row offset = tap.
// Plain: lda = K, row offset = 2.
// ---------------------------------------------------------------------------
template <bool CONV>
__global__ __launch_bounds__(256) void wmma_gemm_bf16_kernel(
    const unsigned short* __restrict__ A, const unsigned short* __restrict__ B,
    const float* __restrict__ bias, float* __restrict__ C,
    int M, int N, int K, int Cin, int relu)
{
  __shared__ alignas(16) unsigned short As[2][128][40];  // [m][k], 80B rows
  __shared__ alignas(16) unsigned short Bs[2][64][40];   // [n][k]

  const int tid  = threadIdx.x;
  const int lane = tid & 31;
  const int hi   = lane >> 4;          // K sub-stripe / M+8 select
  const int lr   = lane & 15;
  const int wid  = tid >> 5;
  const int wy   = wid & 3;            // 32-row group
  const int wx   = wid >> 2;           // 32-col group

  const int bm = blockIdx.y * 128;
  const int bn = blockIdx.x * 64;
  const int lda = CONV ? Cin : K;

  f32x8 acc00 = {}, acc01 = {}, acc10 = {}, acc11 = {};

  auto issue_tiles = [&](int buf, int colbase, int rowoff, int k0) {
    tdm_tile_load(A + (long)(bm + rowoff) * lda + colbase,
                  (unsigned int)(unsigned long)(void*)&As[buf][0][0],
                  lda, 128);
    tdm_tile_load(B + (long)bn * K + k0,
                  (unsigned int)(unsigned long)(void*)&Bs[buf][0][0],
                  K, 64);
  };

  // ---- pipeline prologue: tile 0 via TDM ---------------------------------
  int tap = 0, ci0 = 0;
  if (wid == 0) {
    issue_tiles(0, CONV ? ci0 : 0, CONV ? tap : 2, 0);
    __builtin_amdgcn_s_wait_tensorcnt(0);
  }

  int buf = 0;
  for (int k0 = 0; k0 < K; k0 += 32) {
    __syncthreads();                       // tile t visible to all waves

    int ntap = tap, nci0 = ci0 + 32;
    if (CONV && nci0 == Cin) { nci0 = 0; ++ntap; }
    const bool have_next = (k0 + 32) < K;
    if (have_next && wid == 0) {
      issue_tiles(buf ^ 1, CONV ? nci0 : (k0 + 32), CONV ? ntap : 2, k0 + 32);
    }

    // fragments: two contiguous 16B LDS chunks each (ds_load_b128)
    const u16x8* arow0 = (const u16x8*)As[buf][wy * 32 + lr];
    const u16x8* arow1 = (const u16x8*)As[buf][wy * 32 + 16 + lr];
    const u16x8* brow0 = (const u16x8*)Bs[buf][wx * 32 + lr];
    const u16x8* brow1 = (const u16x8*)Bs[buf][wx * 32 + 16 + lr];
    BF16Frag a0, a1, b0, b1;
    a0.h[0] = arow0[hi]; a0.h[1] = arow0[hi + 2];
    a1.h[0] = arow1[hi]; a1.h[1] = arow1[hi + 2];
    b0.h[0] = brow0[hi]; b0.h[1] = brow0[hi + 2];
    b1.h[0] = brow1[hi]; b1.h[1] = brow1[hi + 2];

    acc00 = __builtin_amdgcn_wmma_f32_16x16x32_bf16(false, a0.v, false, b0.v, (short)0, acc00, false, false);
    acc01 = __builtin_amdgcn_wmma_f32_16x16x32_bf16(false, a0.v, false, b1.v, (short)0, acc01, false, false);
    acc10 = __builtin_amdgcn_wmma_f32_16x16x32_bf16(false, a1.v, false, b0.v, (short)0, acc10, false, false);
    acc11 = __builtin_amdgcn_wmma_f32_16x16x32_bf16(false, a1.v, false, b1.v, (short)0, acc11, false, false);

    if (have_next && wid == 0) {
      __builtin_amdgcn_s_wait_tensorcnt(0);  // tile t+1 landed before barrier
    }
    buf ^= 1;
    tap = ntap; ci0 = nci0;
  }

  // ---- epilogue: VGPR r -> M = r (+8 for lane>=16), col = lane&15 --------
  const int c0 = bn + wx * 32 + lr;
  const int c1 = c0 + 16;
  const float bv0 = bias ? bias[c0] : 0.f;
  const float bv1 = bias ? bias[c1] : 0.f;
#pragma unroll
  for (int r = 0; r < 8; ++r) {
    int row0 = bm + wy * 32 + r + hi * 8;
    int row1 = row0 + 16;
    if (row0 < M) {
      float v = acc00[r] + bv0;
      if (relu) v = fmaxf(v, 0.f);
      C[(long)row0 * N + c0] = v;
      v = acc01[r] + bv1;
      if (relu) v = fmaxf(v, 0.f);
      C[(long)row0 * N + c1] = v;
    }
    if (row1 < M) {
      float v = acc10[r] + bv0;
      if (relu) v = fmaxf(v, 0.f);
      C[(long)row1 * N + c0] = v;
      v = acc11[r] + bv1;
      if (relu) v = fmaxf(v, 0.f);
      C[(long)row1 * N + c1] = v;
    }
  }
}

// ------------------------- bf16 pre-conversion passes ----------------------
// f32 [M,C] -> bf16 halo'd [rowsH,C]: logical row m at halo row m+2, rest 0.
__global__ __launch_bounds__(256) void cvt_halo_kernel(
    const float* __restrict__ src, unsigned short* __restrict__ dst,
    int M, int C)
{
  int row = blockIdx.x;
  bool interior = (row >= 2) && (row < M + 2);
  const float* s = src + (long)(row - 2) * C;
  unsigned short* d = dst + (long)row * C;
  for (int c = threadIdx.x; c < C; c += 256)
    d[c] = interior ? (unsigned short)f2bfbits(s[c]) : (unsigned short)0;
}
// W [K,N] f32 -> bf16 transposed [N,K]
__global__ void cvt_w_t_kernel(const float* __restrict__ W,
                               unsigned short* __restrict__ out, int K, int N) {
  long idx = (long)blockIdx.x * 256 + threadIdx.x;
  long total = (long)K * N;
  if (idx >= total) return;
  int n = (int)(idx / K), kk = (int)(idx % K);
  out[idx] = (unsigned short)f2bfbits(W[(long)kk * N + n]);
}
// conv K[N][Cin][5] -> bf16 Bp[n][kk], kk = t*Cin + ci (tap-major).
__global__ void conv_pack_kernel(const float* __restrict__ Kf,
                                 unsigned short* __restrict__ Bp,
                                 int Cin, int N) {
  long idx = (long)blockIdx.x * 256 + threadIdx.x;
  int Kdim = Cin * 5;
  long total = (long)Kdim * N;
  if (idx >= total) return;
  int n = (int)(idx / Kdim), kk = (int)(idx % Kdim);
  int t = kk / Cin, ci = kk % Cin;
  Bp[idx] = (unsigned short)f2bfbits(Kf[((long)n * Cin + ci) * 5 + t]);
}
__global__ void zero_rows_u16_kernel(unsigned short* __restrict__ base,
                                     int startRow, int C) {
  unsigned short* d = base + (long)(startRow + blockIdx.x) * C;
  for (int c = threadIdx.x; c < C; c += 256) d[c] = 0;
}

// ------------------------- GCN propagation ---------------------------------
__global__ void deg_init_kernel(float* deg, int Nn) {
  int i = blockIdx.x * 256 + threadIdx.x;
  if (i < Nn) deg[i] = 1.0f;                        // self loop
}
__global__ void deg_edge_kernel(const int* __restrict__ dst, float* deg, int E) {
  int e = blockIdx.x * 256 + threadIdx.x;
  if (e < E) atomicAdd(&deg[dst[e]], 1.0f);
}
__global__ void dinv_kernel(const float* __restrict__ deg, float* dinv, int Nn) {
  int i = blockIdx.x * 256 + threadIdx.x;
  if (i < Nn) dinv[i] = rsqrtf(deg[i]);
}
__global__ __launch_bounds__(256) void prop_init_kernel(
    const float* __restrict__ hin, const float* __restrict__ dinv,
    float* __restrict__ out, int Nn)
{
  int row = blockIdx.x;
  if (row >= Nn) return;
  float w = dinv[row] * dinv[row];
  long base = (long)row * 512;
  out[base + threadIdx.x]       = hin[base + threadIdx.x] * w;
  out[base + threadIdx.x + 256] = hin[base + threadIdx.x + 256] * w;
}
__global__ __launch_bounds__(256) void prop_edge_kernel(
    const float* __restrict__ hin, float* __restrict__ out,
    const int* __restrict__ src, const int* __restrict__ dst,
    const float* __restrict__ dinv, int E)
{
  int e = blockIdx.x;
  if (e >= E) return;
  int s = src[e], d = dst[e];
  float nrm = dinv[s] * dinv[d];
  long sb = (long)s * 512, db = (long)d * 512;
  atomicAdd(&out[db + threadIdx.x],       hin[sb + threadIdx.x] * nrm);
  atomicAdd(&out[db + threadIdx.x + 256], hin[sb + threadIdx.x + 256] * nrm);
}
__global__ __launch_bounds__(256) void bias_relu_kernel(
    float* __restrict__ data, const float* __restrict__ b, int Nn)
{
  int row = blockIdx.x;
  if (row >= Nn) return;
  long base = (long)row * 512;
  int t = threadIdx.x;
  data[base + t]       = fmaxf(data[base + t]       + b[t],       0.f);
  data[base + t + 256] = fmaxf(data[base + t + 256] + b[t + 256], 0.f);
}
__global__ __launch_bounds__(256) void bias_relu_cvt_kernel(
    const float* __restrict__ in, const float* __restrict__ b,
    unsigned short* __restrict__ out, int M)
{
  int row = blockIdx.x;                 // 0..rowsH-1
  int t = threadIdx.x;
  bool interior = (row >= 2) && (row < M + 2);
  long ob = (long)row * 512;
  if (interior) {
    long ib = (long)(row - 2) * 512;
    out[ob + t] =
        (unsigned short)f2bfbits(fmaxf(in[ib + t] + b[t], 0.f));
    out[ob + t + 256] =
        (unsigned short)f2bfbits(fmaxf(in[ib + t + 256] + b[t + 256], 0.f));
  } else {
    out[ob + t] = 0;
    out[ob + t + 256] = 0;
  }
}

// ------------------------- BatchNorm (+ReLU) -------------------------------
__global__ __launch_bounds__(256) void bn_relu_kernel(
    float* __restrict__ data, const float* __restrict__ gamma,
    const float* __restrict__ beta, int L, int C)
{
  int ch = blockIdx.x;
  __shared__ float sh[256], sh2[256];
  float s = 0.f, q = 0.f;
  for (int l = threadIdx.x; l < L; l += 256) {
    float v = data[(long)l * C + ch];
    s += v; q += v * v;
  }
  sh[threadIdx.x] = s; sh2[threadIdx.x] = q;
  __syncthreads();
  for (int off = 128; off > 0; off >>= 1) {
    if (threadIdx.x < off) {
      sh[threadIdx.x]  += sh[threadIdx.x + off];
      sh2[threadIdx.x] += sh2[threadIdx.x + off];
    }
    __syncthreads();
  }
  float mean = sh[0] / (float)L;
  float var  = sh2[0] / (float)L - mean * mean;
  float inv  = rsqrtf(var + 1e-5f);
  float g = gamma[ch], b = beta[ch];
  for (int l = threadIdx.x; l < L; l += 256) {
    float v = data[(long)l * C + ch];
    v = g * (v - mean) * inv + b;
    data[(long)l * C + ch] = fmaxf(v, 0.f);
  }
}
__global__ __launch_bounds__(256) void bn_relu_cvt_kernel(
    const float* __restrict__ data, const float* __restrict__ gamma,
    const float* __restrict__ beta, unsigned short* __restrict__ out,
    int L, int C)
{
  int ch = blockIdx.x;
  __shared__ float sh[256], sh2[256];
  float s = 0.f, q = 0.f;
  for (int l = threadIdx.x; l < L; l += 256) {
    float v = data[(long)l * C + ch];
    s += v; q += v * v;
  }
  sh[threadIdx.x] = s; sh2[threadIdx.x] = q;
  __syncthreads();
  for (int off = 128; off > 0; off >>= 1) {
    if (threadIdx.x < off) {
      sh[threadIdx.x]  += sh[threadIdx.x + off];
      sh2[threadIdx.x] += sh2[threadIdx.x + off];
    }
    __syncthreads();
  }
  float mean = sh[0] / (float)L;
  float var  = sh2[0] / (float)L - mean * mean;
  float inv  = rsqrtf(var + 1e-5f);
  float g = gamma[ch], b = beta[ch];
  for (int l = threadIdx.x; l < L; l += 256) {
    float v = data[(long)l * C + ch];
    v = g * (v - mean) * inv + b;
    out[(long)(l + 2) * C + ch] = (unsigned short)f2bfbits(fmaxf(v, 0.f));
  }
}

// ------------------------- attention (algebraically reduced) ----------------
__global__ __launch_bounds__(256) void dom_query_kernel(
    const float* __restrict__ dom_table, const int* __restrict__ ids, int nids,
    const float* __restrict__ Wq, const float* __restrict__ bq,
    float* __restrict__ qa)
{
  __shared__ float qs[256];
  int t = threadIdx.x;
  float q = 0.f;
  for (int j = 0; j < nids; ++j) q += dom_table[(long)ids[j] * 256 + t];
  qs[t] = q;
  __syncthreads();
  float acc = bq[t];
  for (int d = 0; d < 256; ++d) acc += qs[d] * Wq[(long)d * 256 + t];
  qa[t] = acc;
}
__global__ __launch_bounds__(256) void attn_u_kernel(
    const float* __restrict__ Wk, const float* __restrict__ bk,
    const float* __restrict__ qa, float* __restrict__ u, float* __restrict__ c0)
{
  __shared__ float qs[256];
  qs[threadIdx.x] = qa[threadIdx.x];
  __syncthreads();
  int p = blockIdx.x * 256 + threadIdx.x;
  float acc = 0.f;
  for (int a = 0; a < 256; ++a) acc += Wk[(long)p * 256 + a] * qs[a];
  u[p] = acc;
  if (blockIdx.x == 0 && threadIdx.x == 0) {
    float c = 0.f;
    for (int a = 0; a < 256; ++a) c += qs[a] * bk[a];
    c0[0] = c;
  }
}
__global__ __launch_bounds__(256) void attn_reset_kernel(
    float* __restrict__ r_acc, float* __restrict__ sumexp,
    unsigned int* __restrict__ maxkey)
{
  r_acc[threadIdx.x] = 0.f;
  r_acc[threadIdx.x + 256] = 0.f;
  if (threadIdx.x == 0) { sumexp[0] = 0.f; *maxkey = 0u; }
}
__global__ __launch_bounds__(256) void attn_scores_kernel(
    const float* __restrict__ feats, const float* __restrict__ u,
    const float* __restrict__ c0p, float* __restrict__ s_out,
    unsigned int* __restrict__ maxkey, int Nn)
{
  int lane = threadIdx.x & 31;
  int wid  = threadIdx.x >> 5;
  int row  = blockIdx.x * 8 + wid;
  if (row >= Nn) return;
  const float* fr = feats + (long)row * 512;
  float acc = 0.f;
  for (int j = lane; j < 512; j += 32) acc += fr[j] * u[j];
  for (int off = 16; off > 0; off >>= 1) acc += __shfl_down(acc, off);
  if (lane == 0) {
    float sc = (acc + c0p[0]) * 0.0625f;            // A^-0.5, A=256
    s_out[row] = sc;
    unsigned int b = __float_as_uint(sc);
    unsigned int key = (b & 0x80000000u) ? ~b : (b | 0x80000000u);
    atomicMax(maxkey, key);
  }
}
__global__ __launch_bounds__(256) void attn_accum_kernel(
    const float* __restrict__ feats, const float* __restrict__ s_in,
    const unsigned int* __restrict__ maxkey, float* __restrict__ r_acc,
    float* __restrict__ sumexp, int Nn)
{
  __shared__ float rp[512];
  rp[threadIdx.x] = 0.f; rp[threadIdx.x + 256] = 0.f;
  __syncthreads();
  unsigned int key = *maxkey;
  unsigned int b = (key & 0x80000000u) ? (key & 0x7FFFFFFFu) : ~key;
  float mx = __uint_as_float(b);
  float se = 0.f;
  for (int row = blockIdx.x; row < Nn; row += gridDim.x) {
    float e = __expf(s_in[row] - mx);
    const float* fr = feats + (long)row * 512;
    rp[threadIdx.x]       += e * fr[threadIdx.x];
    rp[threadIdx.x + 256] += e * fr[threadIdx.x + 256];
    se += e;
  }
  __syncthreads();
  atomicAdd(&r_acc[threadIdx.x], rp[threadIdx.x]);
  atomicAdd(&r_acc[threadIdx.x + 256], rp[threadIdx.x + 256]);
  if (threadIdx.x == 0) atomicAdd(sumexp, se);
}
__global__ __launch_bounds__(256) void attn_emb_kernel(
    const float* __restrict__ r_acc, const float* __restrict__ sumexp,
    const float* __restrict__ Wv, const float* __restrict__ bv,
    float* __restrict__ emb)
{
  __shared__ float rw[512];
  float inv = 1.0f / sumexp[0];
  rw[threadIdx.x]       = r_acc[threadIdx.x] * inv;
  rw[threadIdx.x + 256] = r_acc[threadIdx.x + 256] * inv;
  __syncthreads();
  int p = blockIdx.x * 256 + threadIdx.x;
  float acc = bv[p];
  for (int k = 0; k < 512; ++k) acc += rw[k] * Wv[(long)k * 512 + p];
  emb[p] = acc;
}
__global__ __launch_bounds__(256) void logits_kernel(
    const float* __restrict__ emb, const float* __restrict__ Wg,
    const float* __restrict__ bg, float* __restrict__ outp, int G)
{
  __shared__ float es[512];
  es[threadIdx.x]       = emb[threadIdx.x];
  es[threadIdx.x + 256] = emb[threadIdx.x + 256];
  __syncthreads();
  int gi = blockIdx.x * 256 + threadIdx.x;
  if (gi >= G) return;
  float acc = bg[gi];
  for (int k = 0; k < 512; ++k) acc += es[k] * Wg[(long)k * G + gi];
  outp[gi] = acc;
}

// ---------------------------------------------------------------------------
extern "C" void kernel_launch(void* const* d_in, const int* in_sizes, int n_in,
                              void* d_out, int out_size, void* d_ws, size_t ws_size,
                              hipStream_t stream) {
  const float* x       = (const float*)d_in[0];
  const int*   eidx    = (const int*)  d_in[1];
  const int*   dom_ids = (const int*)  d_in[2];
  const float* W1  = (const float*)d_in[3];
  const float* b1  = (const float*)d_in[4];
  const float* W2  = (const float*)d_in[5];
  const float* b2  = (const float*)d_in[6];
  const float* K1  = (const float*)d_in[7];
  const float* cb1 = (const float*)d_in[8];
  const float* g1  = (const float*)d_in[9];
  const float* be1 = (const float*)d_in[10];
  const float* K2  = (const float*)d_in[11];
  const float* cb2 = (const float*)d_in[12];
  const float* g2  = (const float*)d_in[13];
  const float* be2 = (const float*)d_in[14];
  const float* dom_table = (const float*)d_in[15];
  const float* Wq  = (const float*)d_in[16];
  const float* bq  = (const float*)d_in[17];
  const float* Wk  = (const float*)d_in[18];
  const float* bk  = (const float*)d_in[19];
  const float* Wv  = (const float*)d_in[20];
  const float* bv  = (const float*)d_in[21];
  const float* Wg  = (const float*)d_in[22];
  const float* bg  = (const float*)d_in[23];
  const float* Wc  = (const float*)d_in[24];
  const float* bc  = (const float*)d_in[25];

  const int F_IN = 1280, HID = 512, P = 512, G = 5000;
  const int Nn = in_sizes[0] / F_IN;      // 30000
  const int E  = in_sizes[1] / 2;         // 480000
  const int nids = in_sizes[2];           // 8
  const int* src  = eidx;
  const int* dstp = eidx + E;

  const int gM    = (Nn + 127) / 128;
  const int Mp    = gM * 128;             // rows padded to block grid
  const int rowsH = Mp + 4;               // +2-row halo each side

  float* ws = (float*)d_ws;
  const long NB = (long)Nn * 512;
  float* BUF0 = ws;
  float* BUF1 = ws + NB;
  float* BUF2 = ws + 2 * NB;              // conv1 out f32, then gcn_feats
  float* BUF3 = ws + 3 * NB;              // cnn_feats
  float* deg    = ws + 4 * NB;
  float* dinv   = deg + Nn;
  float* scores = dinv + Nn;
  float* small  = scores + Nn;
  float* qa     = small;
  float* u      = small + 256;
  float* c0     = small + 768;
  unsigned int* maxkey = (unsigned int*)(small + 769);
  float* sumexp = small + 770;
  float* r_acc  = small + 772;
  float* emb    = small + 1288;
  // bf16 region (16B aligned: all prior offsets are multiples of 4 floats)
  unsigned short* XB0 = (unsigned short*)(small + 4096);   // x,     [rowsH,1280]
  unsigned short* XB1 = XB0 + (long)rowsH * F_IN;          // gcn h, [rowsH, 512]
  unsigned short* XB2 = XB1 + (long)rowsH * 512;           // cnn c1,[rowsH, 512]
  unsigned short* W1b = XB2 + (long)rowsH * 512;           // [512,1280] (N,K)
  unsigned short* W2b = W1b + (long)F_IN * HID;            // [512,512]
  unsigned short* K1b = W2b + (long)HID * P;               // [512,6400]
  unsigned short* K2b = K1b + (long)F_IN * 5 * HID;        // [512,2560]

  dim3 blk(256);

  // ---------------- bf16 pre-conversion passes ----------------------------
  cvt_halo_kernel<<<rowsH, blk, 0, stream>>>(x, XB0, Nn, F_IN);
  cvt_w_t_kernel<<<(unsigned)(((long)F_IN * HID + 255) / 256), blk, 0, stream>>>(
      W1, W1b, F_IN, HID);
  cvt_w_t_kernel<<<(unsigned)(((long)HID * P + 255) / 256), blk, 0, stream>>>(
      W2, W2b, HID, P);
  conv_pack_kernel<<<(unsigned)(((long)F_IN * 5 * HID + 255) / 256), blk, 0, stream>>>(
      K1, K1b, F_IN, HID);
  conv_pack_kernel<<<(unsigned)(((long)HID * 5 * P + 255) / 256), blk, 0, stream>>>(
      K2, K2b, HID, P);

  // ---------------- CNN stream (conv-as-im2col WMMA GEMMs) ----------------
  wmma_gemm_bf16_kernel<true><<<dim3(HID / 64, gM), blk, 0, stream>>>(
      XB0, K1b, cb1, BUF2, Nn, HID, F_IN * 5, F_IN, 0);
  bn_relu_cvt_kernel<<<HID, blk, 0, stream>>>(BUF2, g1, be1, XB2, Nn, HID);
  zero_rows_u16_kernel<<<2, blk, 0, stream>>>(XB2, 0, 512);
  zero_rows_u16_kernel<<<rowsH - (Nn + 2), blk, 0, stream>>>(XB2, Nn + 2, 512);
  wmma_gemm_bf16_kernel<true><<<dim3(P / 64, gM), blk, 0, stream>>>(
      XB2, K2b, cb2, BUF3, Nn, P, HID * 5, HID, 0);
  bn_relu_kernel<<<P, blk, 0, stream>>>(BUF3, g2, be2, Nn, P);   // cnn_feats

  // ---------------- degrees (self-loops + in-degree) ----------------------
  deg_init_kernel<<<(Nn + 255) / 256, blk, 0, stream>>>(deg, Nn);
  deg_edge_kernel<<<(E + 255) / 256, blk, 0, stream>>>(dstp, deg, E);
  dinv_kernel<<<(Nn + 255) / 256, blk, 0, stream>>>(deg, dinv, Nn);

  // ---------------- GCN stream --------------------------------------------
  wmma_gemm_bf16_kernel<false><<<dim3(HID / 64, gM), blk, 0, stream>>>(
      XB0, W1b, nullptr, BUF0, Nn, HID, F_IN, 0, 0);
  prop_init_kernel<<<Nn, blk, 0, stream>>>(BUF0, dinv, BUF1, Nn);
  prop_edge_kernel<<<E, blk, 0, stream>>>(BUF0, BUF1, src, dstp, dinv, E);
  bias_relu_cvt_kernel<<<rowsH, blk, 0, stream>>>(BUF1, b1, XB1, Nn);

  wmma_gemm_bf16_kernel<false><<<dim3(P / 64, gM), blk, 0, stream>>>(
      XB1, W2b, nullptr, BUF0, Nn, P, HID, 0, 0);
  prop_init_kernel<<<Nn, blk, 0, stream>>>(BUF0, dinv, BUF2, Nn);
  prop_edge_kernel<<<E, blk, 0, stream>>>(BUF0, BUF2, src, dstp, dinv, E);
  bias_relu_kernel<<<Nn, blk, 0, stream>>>(BUF2, b2, Nn);   // gcn_feats

  // ---------------- domain query + shared u-vector ------------------------
  dom_query_kernel<<<1, blk, 0, stream>>>(dom_table, dom_ids, nids, Wq, bq, qa);
  attn_u_kernel<<<2, blk, 0, stream>>>(Wk, bk, qa, u, c0);

  // ---------------- two attention streams + logits ------------------------
  float* outp = (float*)d_out;
  for (int sidx = 0; sidx < 2; ++sidx) {
    const float* feats = sidx ? BUF3 : BUF2;
    const float* Wl    = sidx ? Wc : Wg;
    const float* bl    = sidx ? bc : bg;
    float* ol = outp + (long)sidx * G;

    attn_reset_kernel<<<1, blk, 0, stream>>>(r_acc, sumexp, maxkey);
    attn_scores_kernel<<<(Nn + 7) / 8, blk, 0, stream>>>(
        feats, u, c0, scores, maxkey, Nn);
    attn_accum_kernel<<<512, blk, 0, stream>>>(
        feats, scores, maxkey, r_acc, sumexp, Nn);
    attn_emb_kernel<<<2, blk, 0, stream>>>(r_acc, sumexp, Wv, bv, emb);
    logits_kernel<<<(G + 255) / 256, blk, 0, stream>>>(emb, Wl, bl, ol, G);
  }
}
